// RelCNN_18674517803444
// MI455X (gfx1250) — matrix operand order repaired and checked by reading, not compiled, not run on hardware
//
#include <hip/hip_runtime.h>

// ---------------------------------------------------------------------------
// RelConv GNN for MI455X (gfx1250, wave32).
//   * Dense projections: V_WMMA_F32_16X16X32_F16 (f16 in, f32 accumulate)
//   * Edge phase (roofline bottleneck): float4 gather + global_atomic_add_f32
// ---------------------------------------------------------------------------

typedef _Float16 half_t;
typedef __attribute__((ext_vector_type(4)))  _Float16 v4h;
typedef __attribute__((ext_vector_type(8)))  _Float16 v8h;
typedef __attribute__((ext_vector_type(16))) _Float16 v16h;
typedef __attribute__((ext_vector_type(8)))  float    v8f;

#define D 64          // feature dim (D_IN == D_OUT)
#define NLAYERS 3

static __device__ __forceinline__ v16h mkfrag(v8h lo, v8h hi) {
    return __builtin_shufflevector(lo, hi, 0, 1, 2, 3, 4, 5, 6, 7,
                                           8, 9, 10, 11, 12, 13, 14, 15);
}

static __device__ __forceinline__ v8f wmma16(v16h a, v16h b, v8f c) {
    // (neg_a, A, neg_b, B, c_mod, C, reuse_a, reuse_b) -- codegen-confirmed form
    return __builtin_amdgcn_wmma_f32_16x16x32_f16(false, a, false, b, (short)0, c,
                                                  false, false);
}

// ---------------------------------------------------------------------------
// Zero-fill (scratch must be re-zeroed every call: graph replay reuses d_ws)
// ---------------------------------------------------------------------------
__global__ void relgcn_zero(float* __restrict__ p, int n) {
    int i = blockIdx.x * blockDim.x + threadIdx.x;
    if (i < n) p[i] = 0.0f;
}

// ---------------------------------------------------------------------------
// f32 -> f16 conversion (features once per call, weights once per call)
// ---------------------------------------------------------------------------
__global__ void relgcn_f2h(const float* __restrict__ s, half_t* __restrict__ d, int n) {
    int i = blockIdx.x * blockDim.x + threadIdx.x;
    if (i < n) d[i] = (half_t)s[i];
}

// ---------------------------------------------------------------------------
// Edge degree counts (layer-invariant). degS[n] = #edges with src==n,
// degD[n] = #edges with dst==n.
// ---------------------------------------------------------------------------
__global__ void relgcn_degree(const int* __restrict__ ei, int E,
                              float* __restrict__ degS, float* __restrict__ degD) {
    int i = blockIdx.x * blockDim.x + threadIdx.x;
    if (i < E) {
        atomicAdd(&degS[ei[i]],     1.0f);
        atomicAdd(&degD[ei[E + i]], 1.0f);
    }
}

// ---------------------------------------------------------------------------
// WMMA fragment loads. Layouts per CDNA5 ISA 7.12.2 (wave32):
//   A (16x32 f16): lane L holds row M=L&15;
//     halves [0..7]  -> K = kb + 8*(L>>4) + j
//     halves [8..15] -> K = kb + 16 + 8*(L>>4) + (j-8)
//   B (32x16 f16): lane L holds col N=L&15;
//     halves [0..7] -> K = kb + 8*(L>>4) + j; [8..15] -> K = kb + 16 + 8*(L>>4) + (j-8)
//   C/D (16x16 f32): lane L, vgpr r -> M = r + 8*(L>>4), N = L&15
// Both fragments are two contiguous 16-byte loads from a row-major f16 row.
// ---------------------------------------------------------------------------
static __device__ __forceinline__ v16h afrag(const half_t* __restrict__ hrow,
                                             int kb, int kh) {
    v8h lo = *(const v8h*)(hrow + kb + 8 * kh);
    v8h hi = *(const v8h*)(hrow + kb + 16 + 8 * kh);
    return mkfrag(lo, hi);
}
static __device__ __forceinline__ v16h bfrag(const half_t* __restrict__ wrow,
                                             int kb, int kh) {
    const half_t* p = wrow + kb + 8 * kh;
    return mkfrag(*(const v8h*)p, *(const v8h*)(p + 16));
}

// ---------------------------------------------------------------------------
// 3-way projection GEMM: P0 = H*W0^T, P1 = H*W1^T, P2 = H*W2^T + b2.
// One wave computes a 16x64 f32 output tile of one matrix (8 WMMAs).
// H, W* are f16 row-major [*, 64]; weights are L0/L2-resident (8KB each).
// ---------------------------------------------------------------------------
__global__ __launch_bounds__(256)
void relgcn_gemm3(const half_t* __restrict__ H,
                  const half_t* __restrict__ W0, const half_t* __restrict__ W1,
                  const half_t* __restrict__ W2, const float* __restrict__ b2,
                  float* __restrict__ P0, float* __restrict__ P1,
                  float* __restrict__ P2, int nTiles) {
    const int wave = threadIdx.x >> 5;
    const int lane = threadIdx.x & 31;
    const int tile = blockIdx.x * 8 + wave;
    if (tile >= nTiles * 3) return;          // uniform per wave: EXEC stays all-1s
    const int m  = tile % 3;
    const int rb = tile / 3;

    const half_t* W  = (m == 0) ? W0 : ((m == 1) ? W1 : W2);
    const int colL   = lane & 15;
    const int kh     = lane >> 4;            // 0 or 1
    const half_t* hrow = H + (size_t)((rb << 4) + colL) * D;
    const half_t* w0 = W + (size_t)(colL +  0) * D;
    const half_t* w1 = W + (size_t)(colL + 16) * D;
    const half_t* w2 = W + (size_t)(colL + 32) * D;
    const half_t* w3 = W + (size_t)(colL + 48) * D;

    v8f a0 = {}, a1 = {}, a2 = {}, a3 = {};
#pragma unroll
    for (int kb = 0; kb < D; kb += 32) {
        const v16h a = afrag(hrow, kb, kh);
        a0 = wmma16(a, bfrag(w0, kb, kh), a0);
        a1 = wmma16(a, bfrag(w1, kb, kh), a1);
        a2 = wmma16(a, bfrag(w2, kb, kh), a2);
        a3 = wmma16(a, bfrag(w3, kb, kh), a3);
    }

    float* P = (m == 0) ? P0 : ((m == 1) ? P1 : P2);
    float bv0 = 0.f, bv1 = 0.f, bv2 = 0.f, bv3 = 0.f;
    if (m == 2) {
        bv0 = b2[colL +  0]; bv1 = b2[colL + 16];
        bv2 = b2[colL + 32]; bv3 = b2[colL + 48];
    }
    const int rbase = (rb << 4) + (kh << 3);
#pragma unroll
    for (int r = 0; r < 8; ++r) {
        float* orow = P + (size_t)(rbase + r) * D + colL;
        orow[ 0] = a0[r] + bv0;
        orow[16] = a1[r] + bv1;
        orow[32] = a2[r] + bv2;
        orow[48] = a3[r] + bv3;
    }
}

// ---------------------------------------------------------------------------
// Edge scatter: A1[dst] += P1[src]; A2[src] += P2[dst].
// 16 threads per edge, float4 (16B) per thread -> coalesced gather,
// global_atomic_add_f32 scatter.
// ---------------------------------------------------------------------------
__global__ __launch_bounds__(256)
void relgcn_scatter(const int* __restrict__ ei, int E,
                    const float* __restrict__ P1, const float* __restrict__ P2,
                    float* __restrict__ A1, float* __restrict__ A2) {
    int t = blockIdx.x * blockDim.x + threadIdx.x;   // E*16 threads total
    if (t >= E * 16) return;
    const int e = t >> 4;
    const int c = (t & 15) << 2;
    const int s = ei[e];
    const int d = ei[E + e];

    const float4 m1 = *(const float4*)(P1 + (size_t)s * D + c);
    float* u1 = A1 + (size_t)d * D + c;
    atomicAdd(u1 + 0, m1.x); atomicAdd(u1 + 1, m1.y);
    atomicAdd(u1 + 2, m1.z); atomicAdd(u1 + 3, m1.w);

    const float4 m2 = *(const float4*)(P2 + (size_t)d * D + c);
    float* u2 = A2 + (size_t)s * D + c;
    atomicAdd(u2 + 0, m2.x); atomicAdd(u2 + 1, m2.y);
    atomicAdd(u2 + 2, m2.z); atomicAdd(u2 + 3, m2.w);
}

// ---------------------------------------------------------------------------
// Combine: h = relu(R + A1/max(degD,1) + A2/max(degS,1)); emit f16 features
// for the next layer's WMMA and the final concat GEMM.
// ---------------------------------------------------------------------------
__global__ __launch_bounds__(256)
void relgcn_combine(const float* __restrict__ R,  const float* __restrict__ A1,
                    const float* __restrict__ A2, const float* __restrict__ degS,
                    const float* __restrict__ degD, half_t* __restrict__ Hout, int N) {
    int t = blockIdx.x * blockDim.x + threadIdx.x;   // N*16 threads
    if (t >= N * 16) return;
    const int n = t >> 4;
    const int c = (t & 15) << 2;
    const float id = 1.0f / fmaxf(degD[n], 1.0f);
    const float is = 1.0f / fmaxf(degS[n], 1.0f);
    const size_t off = (size_t)n * D + c;
    const float4 r = *(const float4*)(R  + off);
    const float4 u = *(const float4*)(A1 + off);
    const float4 v = *(const float4*)(A2 + off);
    v4h o;
    o[0] = (half_t)fmaxf(r.x + u.x * id + v.x * is, 0.0f);
    o[1] = (half_t)fmaxf(r.y + u.y * id + v.y * is, 0.0f);
    o[2] = (half_t)fmaxf(r.z + u.z * id + v.z * is, 0.0f);
    o[3] = (half_t)fmaxf(r.w + u.w * id + v.w * is, 0.0f);
    *(v4h*)(Hout + off) = o;
}

// ---------------------------------------------------------------------------
// Final GEMM: out = concat([x,h1,h2,h3]) @ FW^T + FB.  FWh: f16 [64, 256].
// One wave per 16-row tile; K=256 over 4 segments (32 WMMAs).
// ---------------------------------------------------------------------------
__global__ __launch_bounds__(256)
void relgcn_final(const half_t* __restrict__ Xh, const half_t* __restrict__ XS,
                  const half_t* __restrict__ FWh, const float* __restrict__ FB,
                  float* __restrict__ OUT, int nTiles, int N) {
    const int wave = threadIdx.x >> 5;
    const int lane = threadIdx.x & 31;
    const int tile = blockIdx.x * 8 + wave;
    if (tile >= nTiles) return;

    const int colL = lane & 15;
    const int kh   = lane >> 4;
    const size_t rowoff = (size_t)((tile << 4) + colL) * D;

    const half_t* f0 = FWh + (size_t)(colL +  0) * 256;
    const half_t* f1 = FWh + (size_t)(colL + 16) * 256;
    const half_t* f2 = FWh + (size_t)(colL + 32) * 256;
    const half_t* f3 = FWh + (size_t)(colL + 48) * 256;

    v8f a0 = {}, a1 = {}, a2 = {}, a3 = {};
#pragma unroll
    for (int seg = 0; seg < 4; ++seg) {
        const half_t* S = (seg == 0) ? Xh : (XS + (size_t)(seg - 1) * N * D);
        const half_t* hrow = S + rowoff;
#pragma unroll
        for (int kb2 = 0; kb2 < D; kb2 += 32) {
            const int kb = (seg << 6) + kb2;     // K within concat (weights)
            const v16h a = afrag(hrow, kb2, kh); // K within segment (features)
            a0 = wmma16(a, bfrag(f0, kb, kh), a0);
            a1 = wmma16(a, bfrag(f1, kb, kh), a1);
            a2 = wmma16(a, bfrag(f2, kb, kh), a2);
            a3 = wmma16(a, bfrag(f3, kb, kh), a3);
        }
    }

    const float bv0 = FB[colL +  0], bv1 = FB[colL + 16];
    const float bv2 = FB[colL + 32], bv3 = FB[colL + 48];
    const int rbase = (tile << 4) + (kh << 3);
#pragma unroll
    for (int r = 0; r < 8; ++r) {
        float* orow = OUT + (size_t)(rbase + r) * D + colL;
        orow[ 0] = a0[r] + bv0;
        orow[16] = a1[r] + bv1;
        orow[32] = a2[r] + bv2;
        orow[48] = a3[r] + bv3;
    }
}

// ---------------------------------------------------------------------------
// Host launcher (graph-capture safe: kernel launches only, all on `stream`)
// ---------------------------------------------------------------------------
extern "C" void kernel_launch(void* const* d_in, const int* in_sizes, int n_in,
                              void* d_out, int out_size, void* d_ws, size_t ws_size,
                              hipStream_t stream) {
    const float* x     = (const float*)d_in[0];
    const int*   ei    = (const int*)  d_in[1];
    const float* lin1  = (const float*)d_in[2];
    const float* lin2  = (const float*)d_in[3];
    const float* rootw = (const float*)d_in[4];
    const float* rootb = (const float*)d_in[5];
    const float* fw    = (const float*)d_in[6];
    const float* fb    = (const float*)d_in[7];
    float* out = (float*)d_out;

    const int N      = in_sizes[0] / D;     // 50000 (multiple of 16)
    const int E      = in_sizes[1] / 2;     // 800000
    const int nTiles = N / 16;

    // Workspace layout:
    //   f32: P1 P2 R A1 A2 (nd each) | degS degD (N each)
    //   f16: Xh (nd) | XS (3*nd) | WhL1 WhL2 WhR (3*D*D each) | FWh (4*D*D)
    float* ws = (float*)d_ws;
    const size_t nd = (size_t)N * D;
    float* P1   = ws;
    float* P2   = P1 + nd;
    float* R    = P2 + nd;
    float* A1   = R  + nd;
    float* A2   = A1 + nd;                  // A1,A2 contiguous for one zero pass
    float* degS = A2 + nd;                  // degS,degD contiguous
    float* degD = degS + N;

    size_t f32cnt = 5 * nd + 2 * (size_t)N;
    f32cnt = (f32cnt + 15) & ~(size_t)15;   // 64B-align the f16 region
    half_t* Xh   = (half_t*)(ws + f32cnt);
    half_t* XS   = Xh + nd;                 // 3 slabs of nd
    half_t* WhL1 = XS + 3 * nd;
    half_t* WhL2 = WhL1 + 3 * D * D;
    half_t* WhR  = WhL2 + 3 * D * D;
    half_t* FWh  = WhR  + 3 * D * D;

    // One-time (per call) conversions: features + all weights to f16
    relgcn_f2h<<<(int)((nd + 255) / 256), 256, 0, stream>>>(x, Xh, (int)nd);
    relgcn_f2h<<<(3 * D * D + 255) / 256, 256, 0, stream>>>(lin1,  WhL1, 3 * D * D);
    relgcn_f2h<<<(3 * D * D + 255) / 256, 256, 0, stream>>>(lin2,  WhL2, 3 * D * D);
    relgcn_f2h<<<(3 * D * D + 255) / 256, 256, 0, stream>>>(rootw, WhR,  3 * D * D);
    relgcn_f2h<<<(4 * D * D + 255) / 256, 256, 0, stream>>>(fw,    FWh,  4 * D * D);

    // Degrees (layer-invariant)
    relgcn_zero<<<(2 * N + 255) / 256, 256, 0, stream>>>(degS, 2 * N);
    relgcn_degree<<<(E + 255) / 256, 256, 0, stream>>>(ei, E, degS, degD);

    const half_t* H = Xh;
    for (int l = 0; l < NLAYERS; ++l) {
        relgcn_zero<<<(int)((2 * nd + 255) / 256), 256, 0, stream>>>(A1, (int)(2 * nd));
        relgcn_gemm3<<<(nTiles * 3 + 7) / 8, 256, 0, stream>>>(
            H, WhL1 + l * D * D, WhL2 + l * D * D, WhR + l * D * D, rootb + l * D,
            P1, P2, R, nTiles);
        relgcn_scatter<<<(E * 16 + 255) / 256, 256, 0, stream>>>(ei, E, P1, P2, A1, A2);
        half_t* Hn = XS + (size_t)l * nd;
        relgcn_combine<<<(N * 16 + 255) / 256, 256, 0, stream>>>(R, A1, A2, degS, degD,
                                                                 Hn, N);
        H = Hn;
    }

    relgcn_final<<<(nTiles + 7) / 8, 256, 0, stream>>>(Xh, XS, FWh, fb, out, nTiles, N);
}
